// DecoderLayer_50749333570112
// MI455X (gfx1250) — matrix operand order repaired and checked by reading, not compiled, not run on hardware
//
#include <hip/hip_runtime.h>
#include <cstdint>
#include <cstddef>

// ---------------- problem constants ----------------
#define S_LEN  1024
#define BATCH  4
#define DMODEL 1024
#define NHEAD  16
#define HDIM   64
#define FFDIM  4096
#define MTOK   (S_LEN*BATCH)   // 4096 token rows
#define ROT_D  32

// flip to 0 if the assembler rejects the async-to-LDS mnemonic
#define USE_ASYNC_LDS 1

typedef __attribute__((ext_vector_type(16))) __bf16 v16bf;
typedef __attribute__((ext_vector_type(8)))  float  v8f;
typedef __attribute__((ext_vector_type(4)))  unsigned int v4u;
typedef __attribute__((ext_vector_type(2)))  unsigned int v2u;

union Frag16 { v16bf bf; v4u q[2]; };   // 16 bf16 = 8 VGPRs = 2x16B

static __device__ __forceinline__ v8f wmma_bf16(v16bf a, v16bf b, v8f c) {
  // D = A(16x32 bf16) * B(32x16 bf16) + C(16x16 f32)
  return __builtin_amdgcn_wmma_f32_16x16x32_bf16(false, a, false, b, (short)0, c,
                                                 false, false);
}

// ---- CDNA5 async global->LDS staging (ASYNCcnt-tracked, bypasses VGPRs) ----
// Generic pointers to __shared__ have the LDS byte offset in their low 32 bits
// (see ISA 10.2, LDS aperture).
static __device__ __forceinline__ void stage16(void* lds, const void* g) {
#if USE_ASYNC_LDS
  unsigned lds_addr = (unsigned)(size_t)lds;
  unsigned long long gaddr = (unsigned long long)(size_t)g;
  asm volatile("global_load_async_to_lds_b128 %0, %1, off"
               :: "v"(lds_addr), "v"(gaddr) : "memory");
#else
  *reinterpret_cast<v4u*>(lds) = *reinterpret_cast<const v4u*>(g);
#endif
}
static __device__ __forceinline__ void stage_wait() {
#if USE_ASYNC_LDS
  asm volatile("s_wait_asynccnt 0x0" ::: "memory");
#endif
}

// ---------------- f32 -> bf16 convert (plain) ----------------
__global__ __launch_bounds__(256) void cvt_f32_bf16(const float* __restrict__ in,
                                                    __bf16* __restrict__ out,
                                                    size_t n) {
  size_t i = ((size_t)blockIdx.x * 256 + threadIdx.x) * 4;
  if (i + 3 < n) {
    float4 v = *reinterpret_cast<const float4*>(in + i);
    out[i + 0] = (__bf16)v.x;
    out[i + 1] = (__bf16)v.y;
    out[i + 2] = (__bf16)v.z;
    out[i + 3] = (__bf16)v.w;
  }
}

// ---------------- f32[K][N] -> bf16[N][K] tiled transpose-convert ----------------
__global__ __launch_bounds__(256) void transpose_cvt_k(const float* __restrict__ in,
                                                       __bf16* __restrict__ out,
                                                       int K, int N) {
  __shared__ __bf16 T[32][36];
  const int t = threadIdx.x;
  const int k0 = blockIdx.y * 32, n0 = blockIdx.x * 32;
  {
    const int r = t >> 3, c4 = (t & 7) * 4;
    float4 v = *reinterpret_cast<const float4*>(in + (size_t)(k0 + r) * N + n0 + c4);
    T[r][c4 + 0] = (__bf16)v.x; T[r][c4 + 1] = (__bf16)v.y;
    T[r][c4 + 2] = (__bf16)v.z; T[r][c4 + 3] = (__bf16)v.w;
  }
  __syncthreads();
  {
    const int r = t >> 3, c4 = (t & 7) * 4;   // r: n-local, c4: k-local
    union { __bf16 a[4]; v2u q; } u;
    #pragma unroll
    for (int j = 0; j < 4; ++j) u.a[j] = T[c4 + j][r];
    *reinterpret_cast<v2u*>(out + (size_t)(n0 + r) * K + k0 + c4) = u.q;
  }
}

// ---------------- RMSNorm: f32 row -> bf16 row ----------------
__global__ __launch_bounds__(256) void rmsnorm_k(const float* __restrict__ x,
                                                 const float* __restrict__ g,
                                                 __bf16* __restrict__ out) {
  const int row = blockIdx.x;
  const float* xr = x + (size_t)row * DMODEL;
  __shared__ float red[8];
  float ss = 0.f;
  #pragma unroll
  for (int c = threadIdx.x; c < DMODEL; c += 256) {
    float v = xr[c];
    ss += v * v;
  }
  #pragma unroll
  for (int m = 16; m >= 1; m >>= 1) ss += __shfl_xor(ss, m, 32);
  if ((threadIdx.x & 31) == 0) red[threadIdx.x >> 5] = ss;
  __syncthreads();
  float tot = 0.f;
  #pragma unroll
  for (int i = 0; i < 8; ++i) tot += red[i];
  const float rms = sqrtf(tot) * (1.0f / 32.0f);   // * D^-0.5, D=1024
  const float inv = 1.0f / (rms + 1e-8f);
  for (int c = threadIdx.x; c < DMODEL; c += 256) {
    out[(size_t)row * DMODEL + c] = (__bf16)(g[c] * xr[c] * inv);
  }
}

// ---------------- tiled WMMA GEMM ----------------
// C[M,N] = A[M,K](bf16) @ Bt[N,K]^T(bf16) + bias [+resid] [silu]
// block tile 128x128, K-step 32, 8 waves (2x4), per-wave 4x2 16x16 frags.
// Double-buffered LDS, async global->LDS staging, ONE barrier per K-step.
// LDS rows padded to 40 elems (20 dwords) -> b128 frag loads bank-conflict-free.
template<bool OUT_BF16, bool SILU, bool RESID>
__global__ __launch_bounds__(256) void gemm_bf16_k(const __bf16* __restrict__ A,
                                                   const __bf16* __restrict__ Bt,
                                                   const float* __restrict__ bias,
                                                   const float* __restrict__ resid,
                                                   void* __restrict__ Cout,
                                                   int M, int N, int K) {
  __shared__ __bf16 As[2][128 * 40];
  __shared__ __bf16 Bs[2][128 * 40];

  const int tid  = threadIdx.x;
  const int wave = tid >> 5;
  const int lane = tid & 31;
  const int half = lane >> 4;
  const int lp   = lane & 15;
  const int wm   = wave >> 2;     // 0..1  (64 rows each)
  const int wn   = wave & 3;      // 0..3  (32 cols each)
  const int m0   = blockIdx.y * 128;
  const int n0   = blockIdx.x * 128;

  auto stage = [&](int buf, int k0) {
    #pragma unroll
    for (int r = 0; r < 2; ++r) {
      const int g = tid + r * 256;
      const int row = g >> 2, c8 = (g & 3) * 8;
      stage16(&As[buf][row * 40 + c8], A + (size_t)(m0 + row) * K + k0 + c8);
      stage16(&Bs[buf][row * 40 + c8], Bt + (size_t)(n0 + row) * K + k0 + c8);
    }
  };

  const v8f zero = {0.f, 0.f, 0.f, 0.f, 0.f, 0.f, 0.f, 0.f};
  v8f c[4][2];
  #pragma unroll
  for (int i = 0; i < 4; ++i)
    #pragma unroll
    for (int j = 0; j < 2; ++j) c[i][j] = zero;

  stage(0, 0);
  stage_wait();
  __syncthreads();

  int cur = 0;
  for (int k0 = 0; k0 < K; k0 += 32, cur ^= 1) {
    if (k0 + 32 < K) stage(cur ^ 1, k0 + 32);   // prefetch next tile async

    Frag16 af[4], bfr[2];
    #pragma unroll
    for (int i = 0; i < 4; ++i) {
      const int base = (wm * 64 + i * 16 + lp) * 40 + half * 8;
      af[i].q[0] = *reinterpret_cast<const v4u*>(&As[cur][base]);
      af[i].q[1] = *reinterpret_cast<const v4u*>(&As[cur][base + 16]);
    }
    #pragma unroll
    for (int j = 0; j < 2; ++j) {
      const int base = (wn * 32 + j * 16 + lp) * 40 + half * 8;
      bfr[j].q[0] = *reinterpret_cast<const v4u*>(&Bs[cur][base]);
      bfr[j].q[1] = *reinterpret_cast<const v4u*>(&Bs[cur][base + 16]);
    }
    #pragma unroll
    for (int i = 0; i < 4; ++i)
      #pragma unroll
      for (int j = 0; j < 2; ++j)
        c[i][j] = wmma_bf16(af[i].bf, bfr[j].bf, c[i][j]);

    stage_wait();       // own async loads for next buffer landed
    __syncthreads();    // all waves done reading cur / writing next
  }

  // ---- epilogue ----
  #pragma unroll
  for (int i = 0; i < 4; ++i) {
    #pragma unroll
    for (int j = 0; j < 2; ++j) {
      #pragma unroll
      for (int v = 0; v < 8; ++v) {
        const int row = m0 + wm * 64 + i * 16 + half * 8 + v;
        const int col = n0 + wn * 32 + j * 16 + lp;
        float val = c[i][j][v] + bias[col];
        if (SILU) val = val / (1.0f + __expf(-val));
        if (RESID) val += resid[(size_t)row * N + col];
        if (OUT_BF16)
          reinterpret_cast<__bf16*>(Cout)[(size_t)row * N + col] = (__bf16)val;
        else
          reinterpret_cast<float*>(Cout)[(size_t)row * N + col] = val;
      }
    }
  }
}

// ---------------- RoPE + split heads relayout (Q, K) ----------------
// in:  [S*B, D] bf16 token-major.  out: [B,H,S,HK] bf16.
__global__ __launch_bounds__(256) void rope_split_k(const __bf16* __restrict__ in,
                                                    __bf16* __restrict__ out) {
  const size_t idx = (size_t)blockIdx.x * 256 + threadIdx.x;
  if (idx >= (size_t)MTOK * DMODEL) return;
  const int hk = (int)(idx % HDIM);
  const int s  = (int)((idx / HDIM) % S_LEN);
  const int h  = (int)((idx / ((size_t)HDIM * S_LEN)) % NHEAD);
  const int b  = (int)(idx / ((size_t)HDIM * S_LEN * NHEAD));
  const __bf16* row = in + ((size_t)s * BATCH + b) * DMODEL + h * HDIM;
  float val;
  if (hk < ROT_D) {
    const int pair = hk >> 1;
    const float inv_freq = __powf(10000.0f, -(float)(2 * pair) / (float)ROT_D);
    const float ang = (float)s * inv_freq;
    const float cs = __cosf(ang), sn = __sinf(ang);
    const float e = (float)row[pair * 2], o = (float)row[pair * 2 + 1];
    val = (hk & 1) ? (o * cs + e * sn) : (e * cs - o * sn);
  } else {
    val = (float)row[hk];
  }
  out[idx] = (__bf16)val;
}

// ---------------- V relayout, transposed per head ----------------
// in: [S*B, D] bf16 token-major.  out: [B*H][HK][S] bf16 (column-major per head)
__global__ __launch_bounds__(256) void vt_relayout_k(const __bf16* __restrict__ in,
                                                     __bf16* __restrict__ out) {
  __shared__ __bf16 T[64][72];
  const int t  = threadIdx.x;
  const int s0 = blockIdx.x * 64;
  const int bh = blockIdx.y;
  const int b  = bh / NHEAD, h = bh % NHEAD;
  #pragma unroll
  for (int r = 0; r < 2; ++r) {
    const int g = t + r * 256;
    const int sl = g >> 3, c8 = (g & 7) * 8;
    *reinterpret_cast<v4u*>(&T[sl][c8]) =
        *reinterpret_cast<const v4u*>(in + ((size_t)(s0 + sl) * BATCH + b) * DMODEL +
                                      h * HDIM + c8);
  }
  __syncthreads();
  #pragma unroll
  for (int r = 0; r < 2; ++r) {
    const int g = t + r * 256;
    const int hk = g >> 3, sc8 = (g & 7) * 8;
    union { __bf16 a[8]; v4u q; } u;
    #pragma unroll
    for (int j = 0; j < 8; ++j) u.a[j] = T[sc8 + j][hk];
    *reinterpret_cast<v4u*>(out + ((size_t)bh * HDIM + hk) * S_LEN + s0 + sc8) = u.q;
  }
}

// ---------------- flash attention (WMMA, online softmax) ----------------
// Qh/Kh: [B*H, S, HK] bf16.  Vt: [B*H, HK, S] bf16.  outTok: [S*B, D] bf16.
// grid: (S/128, B*H); 8 waves, each owns a 16-row q tile; KV chunks of 32
// double-buffered in LDS via async loads; one barrier per chunk.
template<bool CAUSAL>
__global__ __launch_bounds__(256) void attn_k(const __bf16* __restrict__ Qh,
                                              const __bf16* __restrict__ Kh,
                                              const __bf16* __restrict__ Vt,
                                              __bf16* __restrict__ outTok) {
  __shared__ __bf16 Ks[2][32 * 72];   // [kv][hk], padded
  __shared__ __bf16 Vs[2][64 * 40];   // [hk][kv], padded
  __shared__ __bf16 Ps[8][16 * 40];   // per-wave P tile, padded

  const int bh   = blockIdx.y;
  const int b    = bh / NHEAD;
  const int h    = bh % NHEAD;
  const int tid  = threadIdx.x;
  const int wave = tid >> 5;
  const int lane = tid & 31;
  const int half = lane >> 4;
  const int lp   = lane & 15;
  const int qbase = blockIdx.x * 128 + wave * 16;

  const __bf16* Q = Qh + (size_t)bh * S_LEN * HDIM;
  const __bf16* K = Kh + (size_t)bh * S_LEN * HDIM;
  const __bf16* V = Vt + (size_t)bh * HDIM * S_LEN;

  auto stage = [&](int buf, int cchunk) {
    {  // K chunk [32 kv][64 hk]
      const int r = tid >> 3, c8 = (tid & 7) * 8;
      stage16(&Ks[buf][r * 72 + c8], &K[(size_t)(cchunk * 32 + r) * HDIM + c8]);
    }
    {  // V chunk transposed [64 hk][32 kv]
      const int r = tid >> 2, c8 = (tid & 3) * 8;
      stage16(&Vs[buf][r * 40 + c8], &V[(size_t)r * S_LEN + cchunk * 32 + c8]);
    }
  };

  // Q fragments for hk 0..31 and 32..63 (A layout, 2x16B each)
  Frag16 qf[2];
  #pragma unroll
  for (int kc = 0; kc < 2; ++kc) {
    const size_t base = (size_t)(qbase + lp) * HDIM + kc * 32 + half * 8;
    qf[kc].q[0] = *reinterpret_cast<const v4u*>(Q + base);
    qf[kc].q[1] = *reinterpret_cast<const v4u*>(Q + base + 16);
  }

  const v8f zero = {0.f, 0.f, 0.f, 0.f, 0.f, 0.f, 0.f, 0.f};
  v8f o[4];
  o[0] = zero; o[1] = zero; o[2] = zero; o[3] = zero;
  float mrow[8], lrow[8];
  #pragma unroll
  for (int v = 0; v < 8; ++v) { mrow[v] = -1e30f; lrow[v] = 0.f; }

  const int nchunks = CAUSAL ? (blockIdx.x * 128 + 128) / 32 : S_LEN / 32;
  stage(0, 0);
  stage_wait();
  __syncthreads();

  int cur = 0;
  for (int cchunk = 0; cchunk < nchunks; ++cchunk, cur ^= 1) {
    if (cchunk + 1 < nchunks) stage(cur ^ 1, cchunk + 1);

    // scores S = Q K^T (16 q rows x 32 kv cols): 2 n-chunks x 2 k-chunks
    v8f sc[2];
    sc[0] = zero; sc[1] = zero;
    #pragma unroll
    for (int j = 0; j < 2; ++j) {
      #pragma unroll
      for (int kc = 0; kc < 2; ++kc) {
        Frag16 kf;
        const int base = (j * 16 + lp) * 72 + kc * 32 + half * 8;
        kf.q[0] = *reinterpret_cast<const v4u*>(&Ks[cur][base]);
        kf.q[1] = *reinterpret_cast<const v4u*>(&Ks[cur][base + 16]);
        sc[j] = wmma_bf16(qf[kc].bf, kf.bf, sc[j]);
      }
    }

    // scale + mask + online softmax
    float s0[8], s1[8];
    #pragma unroll
    for (int v = 0; v < 8; ++v) {
      float x0 = sc[0][v] * 0.125f;
      float x1 = sc[1][v] * 0.125f;
      if (CAUSAL) {
        const int qr = qbase + half * 8 + v;
        if (cchunk * 32 + lp > qr)      x0 = -1e30f;
        if (cchunk * 32 + 16 + lp > qr) x1 = -1e30f;
      }
      s0[v] = x0; s1[v] = x1;
    }
    #pragma unroll
    for (int v = 0; v < 8; ++v) {
      float mx = fmaxf(s0[v], s1[v]);
      #pragma unroll
      for (int d = 8; d >= 1; d >>= 1) mx = fmaxf(mx, __shfl_xor(mx, d, 32));
      const float mnew  = fmaxf(mrow[v], mx);
      const float alpha = __expf(mrow[v] - mnew);
      mrow[v] = mnew;
      const float p0 = __expf(s0[v] - mnew);
      const float p1 = __expf(s1[v] - mnew);
      float rs = p0 + p1;
      #pragma unroll
      for (int d = 8; d >= 1; d >>= 1) rs += __shfl_xor(rs, d, 32);
      lrow[v] = lrow[v] * alpha + rs;
      #pragma unroll
      for (int jj = 0; jj < 4; ++jj) o[jj][v] *= alpha;
      // park P in per-wave LDS (C layout -> row-major 16x32, padded to 40)
      Ps[wave][(half * 8 + v) * 40 + lp]      = (__bf16)p0;
      Ps[wave][(half * 8 + v) * 40 + 16 + lp] = (__bf16)p1;
    }
    asm volatile("s_wait_dscnt 0x0" ::: "memory");

    // reload P in A layout (16 q rows x 32 kv): 2x16B
    Frag16 pa;
    {
      const int base = lp * 40 + half * 8;
      pa.q[0] = *reinterpret_cast<const v4u*>(&Ps[wave][base]);
      pa.q[1] = *reinterpret_cast<const v4u*>(&Ps[wave][base + 16]);
    }
    // O += P V : 4 output n-chunks over hk, V frag = 2x16B from [hk][kv] tile
    #pragma unroll
    for (int j = 0; j < 4; ++j) {
      Frag16 vf;
      const int base = (j * 16 + lp) * 40 + half * 8;
      vf.q[0] = *reinterpret_cast<const v4u*>(&Vs[cur][base]);
      vf.q[1] = *reinterpret_cast<const v4u*>(&Vs[cur][base + 16]);
      o[j] = wmma_bf16(pa.bf, vf.bf, o[j]);
    }

    stage_wait();
    __syncthreads();
  }

  // normalize and write combined heads: out[(s*B+b), h*64 + hk]
  #pragma unroll
  for (int j = 0; j < 4; ++j) {
    #pragma unroll
    for (int v = 0; v < 8; ++v) {
      const int qr  = qbase + half * 8 + v;
      const int col = h * HDIM + j * 16 + lp;
      const float val = o[j][v] / lrow[v];
      outTok[((size_t)qr * BATCH + b) * DMODEL + col] = (__bf16)val;
    }
  }
}

// ---------------- host-side orchestration ----------------
extern "C" void kernel_launch(void* const* d_in, const int* in_sizes, int n_in,
                              void* d_out, int out_size, void* d_ws, size_t ws_size,
                              hipStream_t stream) {
  (void)in_sizes; (void)n_in; (void)out_size; (void)ws_size;
  const float* x_in   = (const float*)d_in[0];
  const float* enc    = (const float*)d_in[1];
  // d_in[2] = tgt_mask (causal tril) -- handled structurally in attn_k<true>
  const float* sa_wq = (const float*)d_in[3];  const float* sa_bq = (const float*)d_in[4];
  const float* sa_wk = (const float*)d_in[5];  const float* sa_bk = (const float*)d_in[6];
  const float* sa_wv = (const float*)d_in[7];  const float* sa_bv = (const float*)d_in[8];
  const float* sa_wo = (const float*)d_in[9];  const float* sa_bo = (const float*)d_in[10];
  const float* ca_wq = (const float*)d_in[11]; const float* ca_bq = (const float*)d_in[12];
  const float* ca_wk = (const float*)d_in[13]; const float* ca_bk = (const float*)d_in[14];
  const float* ca_wv = (const float*)d_in[15]; const float* ca_bv = (const float*)d_in[16];
  const float* ca_wo = (const float*)d_in[17]; const float* ca_bo = (const float*)d_in[18];
  const float* ff_w1 = (const float*)d_in[19]; const float* ff_b1 = (const float*)d_in[20];
  const float* ff_w2 = (const float*)d_in[21]; const float* ff_b2 = (const float*)d_in[22];
  const float* g1 = (const float*)d_in[23];
  const float* g2 = (const float*)d_in[24];
  const float* g3 = (const float*)d_in[25];
  float* out = (float*)d_out;

  // workspace carve-up
  size_t off = 0;
  auto carve = [&](size_t bytes) -> void* {
    void* p = (char*)d_ws + off;
    off += (bytes + 255) & ~(size_t)255;
    return p;
  };
  const size_t DD = (size_t)DMODEL * DMODEL;
  const size_t MD = (size_t)MTOK * DMODEL;
  // transposed bf16 weights: wt[N][K]
  __bf16* w_saq = (__bf16*)carve(DD * 2);  __bf16* w_sak = (__bf16*)carve(DD * 2);
  __bf16* w_sav = (__bf16*)carve(DD * 2);  __bf16* w_sao = (__bf16*)carve(DD * 2);
  __bf16* w_caq = (__bf16*)carve(DD * 2);  __bf16* w_cak = (__bf16*)carve(DD * 2);
  __bf16* w_cav = (__bf16*)carve(DD * 2);  __bf16* w_cao = (__bf16*)carve(DD * 2);
  __bf16* w_ff1 = (__bf16*)carve((size_t)DMODEL * FFDIM * 2);
  __bf16* w_ff2 = (__bf16*)carve((size_t)FFDIM * DMODEL * 2);
  __bf16* enc_bf = (__bf16*)carve(MD * 2);
  __bf16* xn_bf  = (__bf16*)carve(MD * 2);
  __bf16* qb = (__bf16*)carve(MD * 2);
  __bf16* kb = (__bf16*)carve(MD * 2);
  __bf16* vbuf = (__bf16*)carve(MD * 2);
  __bf16* Qh = (__bf16*)carve(MD * 2);
  __bf16* Kh = (__bf16*)carve(MD * 2);
  __bf16* Vt = (__bf16*)carve(MD * 2);
  __bf16* attn_bf = (__bf16*)carve(MD * 2);
  float*  x1 = (float*)carve(MD * 4);
  float*  x2 = (float*)carve(MD * 4);
  __bf16* hbuf = (__bf16*)carve((size_t)MTOK * FFDIM * 2);

  auto tcvt = [&](const float* src, __bf16* dst, int K, int N) {
    transpose_cvt_k<<<dim3(N / 32, K / 32), 256, 0, stream>>>(src, dst, K, N);
  };
  // weights -> transposed bf16
  tcvt(sa_wq, w_saq, DMODEL, DMODEL); tcvt(sa_wk, w_sak, DMODEL, DMODEL);
  tcvt(sa_wv, w_sav, DMODEL, DMODEL); tcvt(sa_wo, w_sao, DMODEL, DMODEL);
  tcvt(ca_wq, w_caq, DMODEL, DMODEL); tcvt(ca_wk, w_cak, DMODEL, DMODEL);
  tcvt(ca_wv, w_cav, DMODEL, DMODEL); tcvt(ca_wo, w_cao, DMODEL, DMODEL);
  tcvt(ff_w1, w_ff1, DMODEL, FFDIM);
  tcvt(ff_w2, w_ff2, FFDIM, DMODEL);
  cvt_f32_bf16<<<dim3((unsigned)(MD / 4 / 256)), 256, 0, stream>>>(enc, enc_bf, MD);

  const dim3 gemmDD(DMODEL / 128, MTOK / 128);         // N=1024
  const dim3 gemmDF(FFDIM / 128, MTOK / 128);          // N=4096
  const dim3 ropeGrid((unsigned)(MD / 256));
  const dim3 vtGrid(S_LEN / 64, BATCH * NHEAD);
  const dim3 attnGrid(S_LEN / 128, BATCH * NHEAD);

  // ===== self-attention =====
  rmsnorm_k<<<MTOK, 256, 0, stream>>>(x_in, g1, xn_bf);
  gemm_bf16_k<true, false, false><<<gemmDD, 256, 0, stream>>>(xn_bf, w_saq, sa_bq, nullptr, qb, MTOK, DMODEL, DMODEL);
  gemm_bf16_k<true, false, false><<<gemmDD, 256, 0, stream>>>(xn_bf, w_sak, sa_bk, nullptr, kb, MTOK, DMODEL, DMODEL);
  gemm_bf16_k<true, false, false><<<gemmDD, 256, 0, stream>>>(xn_bf, w_sav, sa_bv, nullptr, vbuf, MTOK, DMODEL, DMODEL);
  rope_split_k<<<ropeGrid, 256, 0, stream>>>(qb, Qh);
  rope_split_k<<<ropeGrid, 256, 0, stream>>>(kb, Kh);
  vt_relayout_k<<<vtGrid, 256, 0, stream>>>(vbuf, Vt);
  attn_k<true><<<attnGrid, 256, 0, stream>>>(Qh, Kh, Vt, attn_bf);
  gemm_bf16_k<false, false, true><<<gemmDD, 256, 0, stream>>>(attn_bf, w_sao, sa_bo, x_in, x1, MTOK, DMODEL, DMODEL);

  // ===== cross-attention =====
  rmsnorm_k<<<MTOK, 256, 0, stream>>>(x1, g2, xn_bf);
  gemm_bf16_k<true, false, false><<<gemmDD, 256, 0, stream>>>(xn_bf, w_caq, ca_bq, nullptr, qb, MTOK, DMODEL, DMODEL);
  gemm_bf16_k<true, false, false><<<gemmDD, 256, 0, stream>>>(enc_bf, w_cak, ca_bk, nullptr, kb, MTOK, DMODEL, DMODEL);
  gemm_bf16_k<true, false, false><<<gemmDD, 256, 0, stream>>>(enc_bf, w_cav, ca_bv, nullptr, vbuf, MTOK, DMODEL, DMODEL);
  rope_split_k<<<ropeGrid, 256, 0, stream>>>(qb, Qh);
  rope_split_k<<<ropeGrid, 256, 0, stream>>>(kb, Kh);
  vt_relayout_k<<<vtGrid, 256, 0, stream>>>(vbuf, Vt);
  attn_k<false><<<attnGrid, 256, 0, stream>>>(Qh, Kh, Vt, attn_bf);
  gemm_bf16_k<false, false, true><<<gemmDD, 256, 0, stream>>>(attn_bf, w_cao, ca_bo, x1, x2, MTOK, DMODEL, DMODEL);

  // ===== feed-forward =====
  rmsnorm_k<<<MTOK, 256, 0, stream>>>(x2, g3, xn_bf);
  gemm_bf16_k<true, true, false><<<gemmDF, 256, 0, stream>>>(xn_bf, w_ff1, ff_b1, nullptr, hbuf, MTOK, FFDIM, DMODEL);
  gemm_bf16_k<false, false, true><<<gemmDD, 256, 0, stream>>>(hbuf, w_ff2, ff_b2, x2, out, MTOK, DMODEL, FFDIM);
}
